// GIN_66245575574017
// MI455X (gfx1250) — matrix operand order repaired and checked by reading, not compiled, not run on hardware
//
#include <hip/hip_runtime.h>
#include <stdint.h>

typedef __attribute__((ext_vector_type(16))) __bf16 v16bf;
typedef __attribute__((ext_vector_type(8)))  __bf16 v8bf;
typedef __attribute__((ext_vector_type(8)))  float  v8f;

__device__ __forceinline__ uint32_t pack2bf(float a, float b) {
    union { __bf16 h[2]; uint32_t u; } p;
    p.h[0] = (__bf16)a;
    p.h[1] = (__bf16)b;
    return p.u;
}

// Packed bf16x2 atomic add at device scope (GLOBAL_ATOMIC_PK_ADD_BF16).
__device__ __forceinline__ void atom_pk_add_bf16(uint32_t* p, uint32_t v) {
    asm volatile("global_atomic_pk_add_bf16 %0, %1, off scope:SCOPE_DEV"
                 :: "v"((uint64_t)(uintptr_t)p), "v"(v)
                 : "memory");
}

// ---------------------------------------------------------------------------
// Zero a buffer in 16-byte units.
// ---------------------------------------------------------------------------
__global__ void zero16_kernel(uint4* __restrict__ p, int n16) {
    int i = blockIdx.x * blockDim.x + threadIdx.x;
    if (i < n16) p[i] = make_uint4(0u, 0u, 0u, 0u);
}

// ---------------------------------------------------------------------------
// f32 -> packed bf16 conversion: out dword i = pack(in[2i], in[2i+1]).
// ---------------------------------------------------------------------------
__global__ void cvt_bf16_kernel(const float* __restrict__ in,
                                uint32_t* __restrict__ out, int ndw) {
    int i = blockIdx.x * blockDim.x + threadIdx.x;
    if (i < ndw) {
        float2 v = *(const float2*)(in + (size_t)i * 2);
        out[i] = pack2bf(v.x, v.y);
    }
}

// ---------------------------------------------------------------------------
// Edge scatter-add on bf16 features: agg[dst] += feat[src].
// One wave per edge; each lane covers 4 bf16 elements = 2 dwords
// -> 2 packed-bf16 atomics per lane, 64 per edge (vs 128 f32 atomics).
// ---------------------------------------------------------------------------
__global__ __launch_bounds__(256)
void scatter_kernel(const uint32_t* __restrict__ featdw,   // bf16 pairs, row = 64 dw
                    const long long* __restrict__ src,
                    const long long* __restrict__ dst,
                    uint32_t* __restrict__ aggdw, int nedges) {
    int e = blockIdx.x * 8 + (threadIdx.x >> 5);
    if (e >= nedges) return;
    int lane = threadIdx.x & 31;
    long long s = src[e];
    long long d = dst[e];
    uint2 v = *(const uint2*)(featdw + (size_t)s * 64 + lane * 2);
    uint32_t* p = aggdw + (size_t)d * 64 + lane * 2;
    atom_pk_add_bf16(p + 0, v.x);
    atom_pk_add_bf16(p + 1, v.y);
}

// ---------------------------------------------------------------------------
// Pack a row-major f32 weight [K=128][ncols] into the WMMA B-operand layout
// for V_WMMA_F32_16X16X32_BF16:
//   per (ktile, ntile): lane L holds col = ntile*16 + L%16,
//   kbase = ktile*32 + (L/16)*16, dword d = bf16 pair (kbase+2d, kbase+2d+1).
// out index = ((kt*ntiles + nt)*32 + lane)*8 + d
// ---------------------------------------------------------------------------
__global__ void pack_kernel(const float* __restrict__ W, int ncols, int ntiles,
                            uint32_t* __restrict__ out) {
    int id = blockIdx.x * 256 + threadIdx.x;
    int total = 4 * ntiles * 256;
    if (id >= total) return;
    int d    = id & 7;
    int lane = (id >> 3) & 31;
    int nt   = (id >> 8) % ntiles;
    int kt   = id / (ntiles * 256);
    int col  = nt * 16 + (lane & 15);
    int k0   = kt * 32 + (lane >> 4) * 16 + 2 * d;
    out[id] = pack2bf(W[(size_t)k0 * ncols + col], W[(size_t)(k0 + 1) * ncols + col]);
}

// ---------------------------------------------------------------------------
// Fused GIN MLP: out = [relu]( relu((x+agg)@Wa + ba) @ Wb + bb )
// Inputs x/agg are bf16 [N][128]; one wave per 16-node tile; hidden tile
// transposed through per-wave LDS.  NT2*16 = layer-B output width.
// OUT_BF16: write bf16 (conv1, feeds next scatter) vs f32 (conv2 logits).
// ---------------------------------------------------------------------------
template <int NT2, bool RELU2, bool OUT_BF16>
__global__ __launch_bounds__(256)
void gin_mlp_kernel(const __bf16* __restrict__ xin,
                    const __bf16* __restrict__ agg,
                    const uint32_t* __restrict__ wapk,
                    const float* __restrict__ ba,
                    const uint32_t* __restrict__ wbpk,
                    const float* __restrict__ bb,
                    void* __restrict__ outv, int ntiles_total) {
    __shared__ __align__(16) __bf16 lds[8][16 * 128];

    const int w    = threadIdx.x >> 5;
    const int lane = threadIdx.x & 31;
    int tile = blockIdx.x * 8 + w;
    if (tile >= ntiles_total) tile = ntiles_total - 1;  // clamp: duplicate work keeps
                                                        // EXEC full + barrier safe
    const int col16 = lane & 15;
    const int kh    = lane >> 4;

    // ---- phase 1: (x + agg) @ Wa, K = 128 in 4 steps of 32 ----------------
    const size_t rowA = (size_t)(tile * 16 + col16) * 128;  // A row = lane%16
    v8f acc[8] = {};

#pragma unroll
    for (int kt = 0; kt < 4; ++kt) {
        const int o1 = kt * 32 + kh * 16;
        v8bf xa = *(const v8bf*)(xin + rowA + o1);
        v8bf xb = *(const v8bf*)(xin + rowA + o1 + 8);
        v8bf ga = *(const v8bf*)(agg + rowA + o1);
        v8bf gb = *(const v8bf*)(agg + rowA + o1 + 8);
        v16bf a;
#pragma unroll
        for (int i = 0; i < 8; ++i) {
            a[i]     = (__bf16)((float)xa[i] + (float)ga[i]);
            a[8 + i] = (__bf16)((float)xb[i] + (float)gb[i]);
        }
#pragma unroll
        for (int n = 0; n < 8; ++n) {
            v16bf b = *(const v16bf*)(wapk + ((size_t)(kt * 8 + n) * 32 + lane) * 8);
            acc[n] = __builtin_amdgcn_wmma_f32_16x16x32_bf16(
                false, a, false, b, (short)0, acc[n], false, false);
        }
    }

    // ---- bias + relu, transpose C-layout -> LDS (row-major bf16 16x128) ---
    __bf16* myl = &lds[w][0];
#pragma unroll
    for (int n = 0; n < 8; ++n) {
        float bias = ba[n * 16 + col16];
#pragma unroll
        for (int r = 0; r < 8; ++r) {
            float v = acc[n][r] + bias;
            v = fmaxf(v, 0.0f);
            myl[(kh * 8 + r) * 128 + n * 16 + col16] = (__bf16)v;
        }
    }
    __syncthreads();

    // ---- phase 2: hidden @ Wb, A operand read back from LDS ---------------
    v8f acc2[NT2] = {};
#pragma unroll
    for (int kt = 0; kt < 4; ++kt) {
        const int o1 = kt * 32 + kh * 16;
        v8bf lo = *(const v8bf*)&myl[col16 * 128 + o1];
        v8bf hi = *(const v8bf*)&myl[col16 * 128 + o1 + 8];
        v16bf a;
#pragma unroll
        for (int i = 0; i < 8; ++i) { a[i] = lo[i]; a[8 + i] = hi[i]; }
#pragma unroll
        for (int n = 0; n < NT2; ++n) {
            v16bf b = *(const v16bf*)(wbpk + ((size_t)(kt * NT2 + n) * 32 + lane) * 8);
            acc2[n] = __builtin_amdgcn_wmma_f32_16x16x32_bf16(
                false, a, false, b, (short)0, acc2[n], false, false);
        }
    }

    // ---- bias (+relu) and store -------------------------------------------
    const int outw = NT2 * 16;
#pragma unroll
    for (int n = 0; n < NT2; ++n) {
        float bias = bb[n * 16 + col16];
#pragma unroll
        for (int r = 0; r < 8; ++r) {
            float v = acc2[n][r] + bias;
            if (RELU2) v = fmaxf(v, 0.0f);
            size_t idx = (size_t)(tile * 16 + kh * 8 + r) * outw + n * 16 + col16;
            if (OUT_BF16) ((__bf16*)outv)[idx] = (__bf16)v;
            else          ((float*)outv)[idx]  = v;
        }
    }
}

// ---------------------------------------------------------------------------
// Row-wise log_softmax over 64 columns. One wave per row, 2 cols per lane.
// ---------------------------------------------------------------------------
__global__ __launch_bounds__(256)
void lsm_kernel(const float* __restrict__ o, float* __restrict__ out, int nrows) {
    int row = blockIdx.x * 8 + (threadIdx.x >> 5);
    if (row >= nrows) return;
    int lane = threadIdx.x & 31;
    const float* p = o + (size_t)row * 64;
    float v0 = p[lane];
    float v1 = p[lane + 32];
    float m = fmaxf(v0, v1);
#pragma unroll
    for (int off = 16; off; off >>= 1) m = fmaxf(m, __shfl_xor(m, off));
    float s = expf(v0 - m) + expf(v1 - m);
#pragma unroll
    for (int off = 16; off; off >>= 1) s += __shfl_xor(s, off);
    float l = m + logf(s);
    float* q = out + (size_t)row * 64;
    q[lane]      = v0 - l;
    q[lane + 32] = v1 - l;
}

// ---------------------------------------------------------------------------
extern "C" void kernel_launch(void* const* d_in, const int* in_sizes, int n_in,
                              void* d_out, int out_size, void* d_ws, size_t ws_size,
                              hipStream_t stream) {
    const float*     x  = (const float*)d_in[0];
    const long long* ei = (const long long*)d_in[1];
    const float* W1 = (const float*)d_in[2];
    const float* b1 = (const float*)d_in[3];
    const float* W2 = (const float*)d_in[4];
    const float* b2 = (const float*)d_in[5];
    const float* W3 = (const float*)d_in[6];
    const float* b3 = (const float*)d_in[7];
    const float* W4 = (const float*)d_in[8];
    const float* b4 = (const float*)d_in[9];

    const int N = in_sizes[0] / 128;   // 50000 nodes
    const int E = in_sizes[1] / 2;     // 800000 edges
    const long long* src = ei;
    const long long* dst = ei + E;

    char* ws = (char*)d_ws;
    size_t off = 0;
    auto carve = [&](size_t bytes) {
        void* p = ws + off;
        off += (bytes + 255) & ~(size_t)255;
        return p;
    };
    uint32_t* xbf   = (uint32_t*)carve((size_t)N * 128 * 2);  // bf16 [N][128]
    uint32_t* aggbf = (uint32_t*)carve((size_t)N * 128 * 2);  // bf16 [N][128]
    uint32_t* h1bf  = (uint32_t*)carve((size_t)N * 128 * 2);  // bf16 [N][128]
    float*    o2    = (float*)carve((size_t)N * 64 * 4);      // f32  [N][64]
    uint32_t* w1pk  = (uint32_t*)carve(4 * 8 * 32 * 8 * 4);
    uint32_t* w2pk  = (uint32_t*)carve(4 * 8 * 32 * 8 * 4);
    uint32_t* w3pk  = (uint32_t*)carve(4 * 8 * 32 * 8 * 4);
    uint32_t* w4pk  = (uint32_t*)carve(4 * 4 * 32 * 8 * 4);

    // Pre-pack weights into WMMA B-operand layout; convert x to bf16.
    pack_kernel<<<32, 256, 0, stream>>>(W1, 128, 8, w1pk);
    pack_kernel<<<32, 256, 0, stream>>>(W2, 128, 8, w2pk);
    pack_kernel<<<32, 256, 0, stream>>>(W3, 128, 8, w3pk);
    pack_kernel<<<16, 256, 0, stream>>>(W4, 64, 4, w4pk);

    const int ndw   = N * 64;                       // bf16-pair dwords per [N][128]
    cvt_bf16_kernel<<<(ndw + 255) / 256, 256, 0, stream>>>(x, xbf, ndw);

    const int z16    = (int)(((size_t)N * 128 * 2) / 16);
    const int zgrid  = (z16 + 255) / 256;
    const int sgrid  = (E + 7) / 8;
    const int ntiles = (N + 15) / 16;               // 3125
    const int ggrid  = (ntiles + 7) / 8;            // 391

    // Conv 1
    zero16_kernel<<<zgrid, 256, 0, stream>>>((uint4*)aggbf, z16);
    scatter_kernel<<<sgrid, 256, 0, stream>>>(xbf, src, dst, aggbf, E);
    gin_mlp_kernel<8, true, true><<<ggrid, 256, 0, stream>>>(
        (const __bf16*)xbf, (const __bf16*)aggbf, w1pk, b1, w2pk, b2, h1bf, ntiles);

    // Conv 2 (input = relu(h1) in bf16)
    zero16_kernel<<<zgrid, 256, 0, stream>>>((uint4*)aggbf, z16);
    scatter_kernel<<<sgrid, 256, 0, stream>>>(h1bf, src, dst, aggbf, E);
    gin_mlp_kernel<4, false, false><<<ggrid, 256, 0, stream>>>(
        (const __bf16*)h1bf, (const __bf16*)aggbf, w3pk, b3, w4pk, b4, o2, ntiles);

    // log_softmax over 64 classes
    lsm_kernel<<<(N + 7) / 8, 256, 0, stream>>>(o2, (float*)d_out, N);
}
